// GCN_13091060318589
// MI455X (gfx1250) — matrix operand order repaired and checked by reading, not compiled, not run on hardware
//
#include <hip/hip_runtime.h>
#include <hip/hip_bf16.h>
#include <math.h>

// ---------------------------------------------------------------------------
// GraphSAGE forward for MI455X (gfx1250, wave32, WMMA).
// GEMMs: v_wmma_f32_16x16x32_f16, K-loop fully unrolled, weights pre-packed
// into per-lane B-fragment layout (one contiguous 32B v16h load per frag).
// SAGE GEMM instantiation is unguarded (M % 16 == 0) -> straight-line loads,
// no per-iteration exec toggling. Edge scatter / pooling: L2-resident float4
// atomic kernels (feature matrix = 25.6 MB << 192 MB L2).
// ---------------------------------------------------------------------------

typedef __attribute__((ext_vector_type(16))) _Float16 v16h;
typedef __attribute__((ext_vector_type(8)))  float    v8f;

#define NN 50000      // nodes
#define NE 800000     // edges
#define NG 500        // graphs
#define FD 128        // F_IN == H == 128
#define CD 10         // classes

// ----------------------------- utility kernels -----------------------------

__global__ void fill_f32(float* __restrict__ p, float v, int n) {
  int i = blockIdx.x * blockDim.x + threadIdx.x;
  if (i < n) p[i] = v;
}

__global__ void degree_kernel(const long long* __restrict__ dst, float* __restrict__ deg, int e) {
  int i = blockIdx.x * blockDim.x + threadIdx.x;
  if (i < e) atomicAdd(&deg[(int)dst[i]], 1.0f);
}

__global__ void count_kernel(const long long* __restrict__ batch, float* __restrict__ cnt, int n) {
  int i = blockIdx.x * blockDim.x + threadIdx.x;
  if (i < n) atomicAdd(&cnt[(int)batch[i]], 1.0f);
}

__global__ void recip_max1(float* __restrict__ p, int n) {
  int i = blockIdx.x * blockDim.x + threadIdx.x;
  if (i < n) p[i] = 1.0f / fmaxf(p[i], 1.0f);
}

// agg[dst] += x[src]  (one wave per edge; lane covers 4 features; float4 gather)
__global__ void scatter_edges(const float* __restrict__ xin,
                              const long long* __restrict__ src,
                              const long long* __restrict__ dstv,
                              float* __restrict__ agg, int e) {
  int idx = blockIdx.x * blockDim.x + threadIdx.x;
  int eidx = idx >> 5;
  if (eidx >= e) return;
  int c = (idx & 31) * 4;
  int s = (int)src[eidx];
  int d = (int)dstv[eidx];
  const float4 v = *(const float4*)(xin + (size_t)s * FD + c);
  float* o = agg + (size_t)d * FD + c;
  atomicAdd(o + 0, v.x);
  atomicAdd(o + 1, v.y);
  atomicAdd(o + 2, v.z);
  atomicAdd(o + 3, v.w);
}

// g[batch[n]] += h[n] * cnt_inv[batch[n]]
__global__ void pool_kernel(const float* __restrict__ h,
                            const long long* __restrict__ batch,
                            const float* __restrict__ cnt_inv,
                            float* __restrict__ g, int n) {
  int idx = blockIdx.x * blockDim.x + threadIdx.x;
  int node = idx >> 5;
  if (node >= n) return;
  int c = (idx & 31) * 4;
  int b = (int)batch[node];
  float s = cnt_inv[b];
  const float4 v = *(const float4*)(h + (size_t)node * FD + c);
  float* o = g + (size_t)b * FD + c;
  atomicAdd(o + 0, v.x * s);
  atomicAdd(o + 1, v.y * s);
  atomicAdd(o + 2, v.z * s);
  atomicAdd(o + 3, v.w * s);
}

// ------------------------------ WMMA fragments ------------------------------
// A (16x32 f16, ISA 7.12.2): lane L holds row L&15;
//   a[0..7]  = K[kb + 8*(L>=16) + i],  a[8..15] = K[kb + 16 + 8*(L>=16) + i]
// B (32x16 f16): lane L holds col L&15; same K order as A
// C (16x16 f32): VGPR j, lane L -> row j + 8*(L>=16), col L&15.
//
// Weights are PRE-PACKED into per-lane B fragments:
//   packed[ ((kbIdx*Ntiles + nt)*32 + lane)*16 + i ]
//     = (f16) W[ (kbIdx*32 + 8*(lane>=16) + (i&7) + 16*(i>>3)) * Ncols + nt*16 + (lane&15) ]
// so the GEMM loads one contiguous v16h (32B) per fragment.

__global__ void pack_w_frag(const float* __restrict__ W, _Float16* __restrict__ out,
                            int K, int Ncols) {
  int idx = blockIdx.x * blockDim.x + threadIdx.x;
  if (idx >= K * Ncols) return;
  int i      = idx & 15;          // half index within lane
  int lane   = (idx >> 4) & 31;
  int t      = idx >> 9;          // t = kbIdx * Ntiles + nt
  int Ntiles = Ncols >> 4;
  int kbIdx  = t / Ntiles;
  int nt     = t - kbIdx * Ntiles;
  int k   = kbIdx * 32 + (lane >> 4) * 8 + (i & 7) + ((i >> 3) * 16);
  int col = nt * 16 + (lane & 15);
  out[idx] = (_Float16)W[(size_t)k * Ncols + col];
}

template<bool GUARD>
__device__ __forceinline__ v16h load_a_frag(const float* __restrict__ A, int row, int M,
                                            int K, int kb, int half_, float scale) {
  v16h a;
  if (!GUARD || row < M) {
    const float* p0 = A + (size_t)row * K + kb + half_ * 8;
    const float* p1 = p0 + 16;
#pragma unroll
    for (int i = 0; i < 8; ++i) {
      a[i]     = (_Float16)(p0[i] * scale);
      a[8 + i] = (_Float16)(p1[i] * scale);
    }
  } else {
#pragma unroll
    for (int i = 0; i < 16; ++i) a[i] = (_Float16)0.0f;
  }
  return a;
}

// out = relu( A1@W1 + (A2*rowscale2)@W2 + bias )   (A2/W2 optional)
// block = 32 * Ntiles threads: wave w = column tile w of 16-row stripe blockIdx.x.
// GUARD=false requires M % 16 == 0 (no row bounds checks emitted).
template<int K, bool GUARD, bool RELU>
__global__ __launch_bounds__(256) void
wmma_gemm_dual(const float* __restrict__ A1, const _Float16* __restrict__ W1p,
               const float* __restrict__ A2, const _Float16* __restrict__ W2p,
               const float* __restrict__ rowscale2,
               const float* __restrict__ bias,
               float* __restrict__ out,
               int M, int Ncols) {
  const int lane   = threadIdx.x & 31;
  const int wave   = threadIdx.x >> 5;
  const int half_  = lane >> 4;
  const int l15    = lane & 15;
  const int Ntiles = Ncols >> 4;
  const int row    = blockIdx.x * 16 + l15;   // A row owned by this lane
  const int col    = wave * 16 + l15;         // B/C column owned by this lane

  const float bv = bias[col];
  v8f c;
#pragma unroll
  for (int j = 0; j < 8; ++j) c[j] = bv;

#pragma unroll
  for (int kb = 0; kb < K; kb += 32) {
    v16h a = load_a_frag<GUARD>(A1, row, M, K, kb, half_, 1.0f);
    v16h b = *(const v16h*)(W1p + (((size_t)(kb >> 5) * Ntiles + wave) << 9) + (lane << 4));
    c = __builtin_amdgcn_wmma_f32_16x16x32_f16(false, a, false, b, (short)0, c, false, false);
  }
  if (A2 != nullptr) {
    const float sc = (!GUARD || row < M) ? rowscale2[row] : 0.0f;
#pragma unroll
    for (int kb = 0; kb < K; kb += 32) {
      v16h a = load_a_frag<GUARD>(A2, row, M, K, kb, half_, sc);
      v16h b = *(const v16h*)(W2p + (((size_t)(kb >> 5) * Ntiles + wave) << 9) + (lane << 4));
      c = __builtin_amdgcn_wmma_f32_16x16x32_f16(false, a, false, b, (short)0, c, false, false);
    }
  }

  const int r0 = blockIdx.x * 16 + half_ * 8;
#pragma unroll
  for (int j = 0; j < 8; ++j) {
    int r = r0 + j;
    if (!GUARD || r < M) {
      float v = c[j];
      if (RELU) v = fmaxf(v, 0.0f);
      out[(size_t)r * Ncols + col] = v;
    }
  }
}

// ------------------------- final layer + log_softmax ------------------------

__global__ void fc3_logsoftmax(const float* __restrict__ g3, const float* __restrict__ w,
                               const float* __restrict__ b, float* __restrict__ out, int Gn) {
  int i = blockIdx.x * blockDim.x + threadIdx.x;
  if (i >= Gn) return;
  const float* r = g3 + (size_t)i * 64;
  float logit[CD];
#pragma unroll
  for (int cc = 0; cc < CD; ++cc) logit[cc] = b[cc];
  for (int k = 0; k < 64; ++k) {
    float v = r[k];
#pragma unroll
    for (int cc = 0; cc < CD; ++cc) logit[cc] += v * w[k * CD + cc];
  }
  float m = logit[0];
#pragma unroll
  for (int cc = 1; cc < CD; ++cc) m = fmaxf(m, logit[cc]);
  float s = 0.0f;
#pragma unroll
  for (int cc = 0; cc < CD; ++cc) s += expf(logit[cc] - m);
  float lse = m + logf(s);
#pragma unroll
  for (int cc = 0; cc < CD; ++cc) out[(size_t)i * CD + cc] = logit[cc] - lse;
}

// --------------------------------- launch ----------------------------------

extern "C" void kernel_launch(void* const* d_in, const int* in_sizes, int n_in,
                              void* d_out, int out_size, void* d_ws, size_t ws_size,
                              hipStream_t stream) {
  (void)in_sizes; (void)n_in; (void)out_size; (void)ws_size;

  const float*     x     = (const float*)d_in[0];
  const long long* ei    = (const long long*)d_in[1];   // int64 per reference
  const long long* batch = (const long long*)d_in[2];
  const float* Wl1 = (const float*)d_in[3];
  const float* Wr1 = (const float*)d_in[4];
  const float* b1  = (const float*)d_in[5];
  const float* Wl2 = (const float*)d_in[6];
  const float* Wr2 = (const float*)d_in[7];
  const float* b2  = (const float*)d_in[8];
  const float* Wl3 = (const float*)d_in[9];
  const float* Wr3 = (const float*)d_in[10];
  const float* b3  = (const float*)d_in[11];
  const float* fc1_w = (const float*)d_in[12];
  const float* fc1_b = (const float*)d_in[13];
  const float* fc2_w = (const float*)d_in[14];
  const float* fc2_b = (const float*)d_in[15];
  const float* fc3_w = (const float*)d_in[16];
  const float* fc3_b = (const float*)d_in[17];

  const long long* src = ei;        // edge_index[0]
  const long long* dst = ei + NE;   // edge_index[1]

  // ----- workspace carve-up (~77.5 MB) -----
  char* p = (char*)d_ws;
  auto carve = [&](size_t bytes) -> char* {
    char* r = p;
    p += (bytes + 255) & ~(size_t)255;
    return r;
  };
  float* deg  = (float*)carve((size_t)NN * 4);           // deg -> deg_inv (in place)
  float* cnt  = (float*)carve((size_t)NG * 4);           // cnt -> cnt_inv (in place)
  float* agg  = (float*)carve((size_t)NN * FD * 4);
  float* hA   = (float*)carve((size_t)NN * FD * 4);
  float* hB   = (float*)carve((size_t)NN * FD * 4);
  float* gbuf = (float*)carve((size_t)NG * FD * 4);      // p1+p2+p3 accumulator
  float* g2   = (float*)carve((size_t)512 * FD * 4);
  float* g3   = (float*)carve((size_t)512 * 64 * 4);
  _Float16* wl1h = (_Float16*)carve((size_t)(7 * 16384 + 8192) * sizeof(_Float16));
  _Float16* wr1h = wl1h + 1 * 16384;
  _Float16* wl2h = wl1h + 2 * 16384;
  _Float16* wr2h = wl1h + 3 * 16384;
  _Float16* wl3h = wl1h + 4 * 16384;
  _Float16* wr3h = wl1h + 5 * 16384;
  _Float16* fc1h = wl1h + 6 * 16384;
  _Float16* fc2h = wl1h + 7 * 16384;

  const int T = 256;
  auto blocks = [](long long n, int t) { return (int)((n + t - 1) / t); };

  // ----- degree / counts / reciprocals -----
  fill_f32<<<blocks(NN, T), T, 0, stream>>>(deg, 0.0f, NN);
  fill_f32<<<blocks(NG, T), T, 0, stream>>>(cnt, 0.0f, NG);
  fill_f32<<<blocks((long long)NG * FD, T), T, 0, stream>>>(gbuf, 0.0f, NG * FD);
  degree_kernel<<<blocks(NE, T), T, 0, stream>>>(dst, deg, NE);
  count_kernel<<<blocks(NN, T), T, 0, stream>>>(batch, cnt, NN);
  recip_max1<<<blocks(NN, T), T, 0, stream>>>(deg, NN);
  recip_max1<<<blocks(NG, T), T, 0, stream>>>(cnt, NG);

  // ----- weights -> f16 B-fragment layout (once per call; L2-resident) -----
  pack_w_frag<<<blocks(16384, T), T, 0, stream>>>(Wl1,   wl1h, FD, FD);
  pack_w_frag<<<blocks(16384, T), T, 0, stream>>>(Wr1,   wr1h, FD, FD);
  pack_w_frag<<<blocks(16384, T), T, 0, stream>>>(Wl2,   wl2h, FD, FD);
  pack_w_frag<<<blocks(16384, T), T, 0, stream>>>(Wr2,   wr2h, FD, FD);
  pack_w_frag<<<blocks(16384, T), T, 0, stream>>>(Wl3,   wl3h, FD, FD);
  pack_w_frag<<<blocks(16384, T), T, 0, stream>>>(Wr3,   wr3h, FD, FD);
  pack_w_frag<<<blocks(16384, T), T, 0, stream>>>(fc1_w, fc1h, FD, FD);
  pack_w_frag<<<blocks(8192,  T), T, 0, stream>>>(fc2_w, fc2h, FD, 64);

  const int MT       = NN / 16;              // 3125 row tiles, exact
  const int GT       = (NG + 15) / 16;       // 32 row tiles (guarded)
  const long long ET = (long long)NE * 32;   // scatter threads
  const long long PT = (long long)NN * 32;   // pool threads

  // ----- layer 1 -----
  fill_f32<<<blocks((long long)NN * FD, T), T, 0, stream>>>(agg, 0.0f, NN * FD);
  scatter_edges<<<blocks(ET, T), T, 0, stream>>>(x, src, dst, agg, NE);
  wmma_gemm_dual<FD, false, true><<<MT, 256, 0, stream>>>(x,  wl1h, agg, wr1h, deg, b1, hA, NN, FD);
  pool_kernel<<<blocks(PT, T), T, 0, stream>>>(hA, batch, cnt, gbuf, NN);

  // ----- layer 2 -----
  fill_f32<<<blocks((long long)NN * FD, T), T, 0, stream>>>(agg, 0.0f, NN * FD);
  scatter_edges<<<blocks(ET, T), T, 0, stream>>>(hA, src, dst, agg, NE);
  wmma_gemm_dual<FD, false, true><<<MT, 256, 0, stream>>>(hA, wl2h, agg, wr2h, deg, b2, hB, NN, FD);
  pool_kernel<<<blocks(PT, T), T, 0, stream>>>(hB, batch, cnt, gbuf, NN);

  // ----- layer 3 -----
  fill_f32<<<blocks((long long)NN * FD, T), T, 0, stream>>>(agg, 0.0f, NN * FD);
  scatter_edges<<<blocks(ET, T), T, 0, stream>>>(hB, src, dst, agg, NE);
  wmma_gemm_dual<FD, false, true><<<MT, 256, 0, stream>>>(hB, wl3h, agg, wr3h, deg, b3, hA, NN, FD);
  pool_kernel<<<blocks(PT, T), T, 0, stream>>>(hA, batch, cnt, gbuf, NN);

  // ----- MLP head (guarded: M = 500) -----
  wmma_gemm_dual<FD, true, true><<<GT, 256, 0, stream>>>(gbuf, fc1h, nullptr, nullptr, nullptr,
                                                         fc1_b, g2, NG, FD);
  wmma_gemm_dual<FD, true, true><<<GT, 128, 0, stream>>>(g2, fc2h, nullptr, nullptr, nullptr,
                                                         fc2_b, g3, NG, 64);
  fc3_logsoftmax<<<blocks(NG, T), T, 0, stream>>>(g3, fc3_w, fc3_b, (float*)d_out, NG);
}
